// DNC_41042707480750
// MI455X (gfx1250) — matrix operand order repaired and checked by reading, not compiled, not run on hardware
//
#include <hip/hip_runtime.h>

typedef __attribute__((ext_vector_type(16))) __bf16 v16bf;
typedef __attribute__((ext_vector_type(8)))  __bf16 v8bf;
typedef __attribute__((ext_vector_type(8)))  float  v8f;

#define B_      64
#define T_      256
#define DIN_    256
#define H_      512
#define NSLOT_  128
#define WW_     64
#define R_      4
#define DOUT_   256
#define KGATES  1024    // [x_t | rv | h]
#define NGATES  2048    // 4*H, gate-interleaved: p = 4*j + gate
#define KIF     512
#define NIF     512     // IFACE padded 465 -> 512
#define IFACE_REAL 465
#define KOUT    768     // [h | rv]
#define NOUT    256
#define KC      256     // K columns staged in LDS per stage
#define AP      264     // LDS pitch for A tile (pad 256+8, breaks bank conflicts)
#define GP      132     // LDS pitch for gates tile (128+4)
#define EPS_    1e-8f

static __device__ __forceinline__ __bf16 f2bf(float f) {
  union { float f; unsigned u; } x; x.f = f;
  unsigned r = (x.u + 0x7FFFu + ((x.u >> 16) & 1u)) >> 16;
  union { unsigned short s; __bf16 b; } y; y.s = (unsigned short)r;
  return y.b;
}
static __device__ __forceinline__ float sigmoidf_(float x) { return 1.f / (1.f + expf(-x)); }
static __device__ __forceinline__ float softplus1_(float x) { return (x > 15.f ? x : log1pf(expf(x))) + 1.f; }

// One wave computes a 16(M) x 64(N) strip: 4 accumulators, K advanced in 32-chunks
// from an LDS-staged A tile (bf16, pitch AP) and row-major bf16 weights W[N][K].
static __device__ __forceinline__ void wmma_strip(
    const __bf16* __restrict__ As, const __bf16* __restrict__ Wt,
    int K, int kc0, int m_blk, int lane, int n0, v8f acc[4])
{
  const int kA  = (lane >> 4) << 4;   // +16 elements for upper-half lanes (A layout)
  const int kB  = (lane >> 4) << 4;   // +16 elements for upper-half lanes (B layout)
  const int row = m_blk * 16 + (lane & 15);
  #pragma unroll
  for (int kk = 0; kk < KC; kk += 32) {
    const __bf16* ap = As + row * AP + kk + kA;
    v8bf h0 = *(const v8bf*)ap;            // K = base..base+7
    v8bf h1 = *(const v8bf*)(ap + 8);      // K = base+8..base+15
    v16bf a;
    #pragma unroll
    for (int i = 0; i < 8; ++i) { a[i] = h0[i]; a[i + 8] = h1[i]; }
    const int kg = kc0 + kk + kB;
    #pragma unroll
    for (int t = 0; t < 4; ++t) {
      const int n = n0 + t * 16 + (lane & 15);
      v16bf b = *(const v16bf*)(Wt + (size_t)n * K + kg);
      acc[t] = __builtin_amdgcn_wmma_f32_16x16x32_bf16(
          false, a, false, b, (short)0, acc[t], false, false);
    }
    __builtin_prefetch(Wt + (size_t)(n0 + (lane & 15)) * K + kg + 32, 0, 1);
  }
}

// ---- gates GEMM + fused LSTM: gates = [x_t|rv|h_in] @ Wcat^T + bias (gate-interleaved),
//      then c,h update for this tile's 32 hidden units via LDS tile. ----
__global__ __launch_bounds__(256) void dnc_gates_lstm(
    const float* __restrict__ x, int t,
    const float* __restrict__ h_in, const float* __restrict__ rv,
    const __bf16* __restrict__ Wcat, const float* __restrict__ bcomb,
    float* __restrict__ c, float* __restrict__ h_out)
{
  extern __shared__ char smem[];
  __bf16* As = (__bf16*)smem;                               // B_*AP bf16 = 33792 B
  float*  Gt = (float*)(smem + (size_t)B_ * AP * 2);        // B_*GP f32 = 33792 B
  const int tid = threadIdx.x, lane = tid & 31, wid = tid >> 5;
  const int m_blk = wid & 3;
  const int n0 = blockIdx.x * 128 + (wid >> 2) * 64;
  v8f z = {0.f,0.f,0.f,0.f,0.f,0.f,0.f,0.f};
  v8f acc[4] = {z, z, z, z};
  for (int kc0 = 0; kc0 < KGATES; kc0 += KC) {
    __syncthreads();
    for (int i = tid; i < B_ * KC; i += 256) {
      int m = i >> 8, kk = i & (KC - 1);
      int kg = kc0 + kk;
      float v;
      if (kg < DIN_)      v = x[(size_t)m * T_ * DIN_ + (size_t)t * DIN_ + kg];
      else if (kg < 512)  v = rv[m * (R_ * WW_) + (kg - DIN_)];
      else                v = h_in[m * H_ + (kg - 512)];
      As[m * AP + kk] = f2bf(v);
    }
    __syncthreads();
    wmma_strip(As, Wcat, KGATES, kc0, m_blk, lane, n0, acc);
  }
  // stash gates tile (with bias) into LDS
  const int pl0 = (wid >> 2) * 64;   // local column base within the 128-wide tile
  #pragma unroll
  for (int tt = 0; tt < 4; ++tt) {
    int nl = pl0 + tt * 16 + (lane & 15);
    int ng = blockIdx.x * 128 + nl;
    #pragma unroll
    for (int r = 0; r < 8; ++r) {
      int m = m_blk * 16 + r + ((lane >> 4) << 3);
      Gt[m * GP + nl] = acc[tt][r] + bcomb[ng];
    }
  }
  __syncthreads();
  // fused LSTM pointwise for this tile's 32 hidden units (columns p = 4*j + gate)
  for (int idx = tid; idx < B_ * 32; idx += 256) {
    int m = idx >> 5, jj = idx & 31;
    int j = blockIdx.x * 32 + jj;
    const float* g4 = &Gt[m * GP + jj * 4];
    float ig = sigmoidf_(g4[0]);
    float fg = sigmoidf_(g4[1]);
    float gg = tanhf(g4[2]);
    float og = sigmoidf_(g4[3]);
    int ci = m * H_ + j;
    float cn = fg * c[ci] + ig * gg;
    c[ci] = cn;
    h_out[ci] = og * tanhf(cn);
  }
}

// ---------------- iface = h @ Wint^T + b_int (padded to 512) ----------------
__global__ __launch_bounds__(256) void dnc_iface_gemm(
    const float* __restrict__ h, const __bf16* __restrict__ Wint,
    const float* __restrict__ bint, float* __restrict__ iface)
{
  extern __shared__ char smem[];
  __bf16* As = (__bf16*)smem;
  const int tid = threadIdx.x, lane = tid & 31, wid = tid >> 5;
  const int m_blk = wid & 3;
  const int n0 = blockIdx.x * 128 + (wid >> 2) * 64;
  v8f z = {0.f,0.f,0.f,0.f,0.f,0.f,0.f,0.f};
  v8f acc[4] = {z, z, z, z};
  for (int kc0 = 0; kc0 < KIF; kc0 += KC) {
    __syncthreads();
    for (int i = tid; i < B_ * KC; i += 256) {
      int m = i >> 8, kk = i & (KC - 1);
      As[m * AP + kk] = f2bf(h[m * H_ + kc0 + kk]);
    }
    __syncthreads();
    wmma_strip(As, Wint, KIF, kc0, m_blk, lane, n0, acc);
  }
  #pragma unroll
  for (int tt = 0; tt < 4; ++tt) {
    int n = n0 + tt * 16 + (lane & 15);
    #pragma unroll
    for (int r = 0; r < 8; ++r) {
      int m = m_blk * 16 + r + ((lane >> 4) << 3);
      iface[(size_t)m * NIF + n] = acc[tt][r] + bint[n];
    }
  }
}

// ---------------- content addressing + memory update + reads ----------------
__global__ __launch_bounds__(128) void dnc_memory_step(
    const float* __restrict__ iface, float* __restrict__ mem, float* __restrict__ rv)
{
  __shared__ float s_if[NIF];
  __shared__ float s_red[NSLOT_];
  __shared__ float s_rv[R_ * WW_];
  __shared__ float s_mem[NSLOT_ * (WW_ + 1)];   // padded pitch 65
  const int b = blockIdx.x, tid = threadIdx.x;
  for (int i = tid; i < NIF; i += 128) s_if[i] = iface[(size_t)b * NIF + i];
  for (int i = tid; i < R_ * WW_; i += 128) s_rv[i] = 0.f;
  __syncthreads();

  float* mrow = mem + ((size_t)b * NSLOT_ + tid) * WW_;
  // --- write content weights (old memory) ---
  float dot = 0.f, nm2 = 0.f, nk2 = 0.f;
  for (int w = 0; w < WW_; ++w) {
    float m = mrow[w], k = s_if[260 + w];
    dot += m * k; nm2 += m * m; nk2 += k * k;
  }
  float nm = fmaxf(sqrtf(nm2), EPS_);
  float nk = fmaxf(sqrtf(nk2), EPS_);
  float wstr = softplus1_(s_if[324]);
  float aa = wstr * dot / (nk * nm);
  s_red[tid] = aa; __syncthreads();
  for (int off = 64; off > 0; off >>= 1) {
    if (tid < off) s_red[tid] = fmaxf(s_red[tid], s_red[tid + off]);
    __syncthreads();
  }
  float amax = s_red[0]; __syncthreads();
  float e = expf(aa - amax);
  s_red[tid] = e; __syncthreads();
  for (int off = 64; off > 0; off >>= 1) {
    if (tid < off) s_red[tid] += s_red[tid + off];
    __syncthreads();
  }
  float ww = e / s_red[0]; __syncthreads();

  // --- memory update; cache new row in LDS ---
  float nm2n = 0.f;
  for (int w = 0; w < WW_; ++w) {
    float er = sigmoidf_(s_if[325 + w]);
    float wv = s_if[389 + w];
    float m = mrow[w];
    m = m * (1.f - ww * er) + ww * wv;
    mrow[w] = m;
    s_mem[tid * (WW_ + 1) + w] = m;
    nm2n += m * m;
  }
  float nmn = fmaxf(sqrtf(nm2n), EPS_);
  __syncthreads();

  // --- read weights on new memory; rv = rw @ mem via LDS atomics ---
  for (int r = 0; r < R_; ++r) {
    float nk2r = 0.f, dr = 0.f;
    for (int w = 0; w < WW_; ++w) {
      float k = s_if[r * WW_ + w];
      nk2r += k * k;
      dr += k * s_mem[tid * (WW_ + 1) + w];
    }
    float nkr = fmaxf(sqrtf(nk2r), EPS_);
    float rstr = softplus1_(s_if[256 + r]);
    float ar = rstr * dr / (nkr * nmn);
    s_red[tid] = ar; __syncthreads();
    for (int off = 64; off > 0; off >>= 1) {
      if (tid < off) s_red[tid] = fmaxf(s_red[tid], s_red[tid + off]);
      __syncthreads();
    }
    float amx = s_red[0]; __syncthreads();
    float er_ = expf(ar - amx);
    s_red[tid] = er_; __syncthreads();
    for (int off = 64; off > 0; off >>= 1) {
      if (tid < off) s_red[tid] += s_red[tid + off];
      __syncthreads();
    }
    float rwn = er_ / s_red[0]; __syncthreads();
    for (int w = 0; w < WW_; ++w)
      atomicAdd(&s_rv[r * WW_ + w], rwn * s_mem[tid * (WW_ + 1) + w]);
  }
  __syncthreads();
  for (int i = tid; i < R_ * WW_; i += 128) rv[(size_t)b * (R_ * WW_) + i] = s_rv[i];
}

// ---------------- final out = [h|rv] @ Wout^T + b_out (last step only) ----------------
__global__ __launch_bounds__(256) void dnc_out_gemm(
    const float* __restrict__ h, const float* __restrict__ rv,
    const __bf16* __restrict__ Wout, const float* __restrict__ bout,
    float* __restrict__ out)
{
  extern __shared__ char smem[];
  __bf16* As = (__bf16*)smem;
  const int tid = threadIdx.x, lane = tid & 31, wid = tid >> 5;
  const int m_blk = wid & 3;
  const int n0 = blockIdx.x * 128 + (wid >> 2) * 64;
  v8f z = {0.f,0.f,0.f,0.f,0.f,0.f,0.f,0.f};
  v8f acc[4] = {z, z, z, z};
  for (int kc0 = 0; kc0 < KOUT; kc0 += KC) {
    __syncthreads();
    for (int i = tid; i < B_ * KC; i += 256) {
      int m = i >> 8, kk = i & (KC - 1);
      int kg = kc0 + kk;
      float v = (kg < H_) ? h[m * H_ + kg] : rv[m * (R_ * WW_) + (kg - H_)];
      As[m * AP + kk] = f2bf(v);
    }
    __syncthreads();
    wmma_strip(As, Wout, KOUT, kc0, m_blk, lane, n0, acc);
  }
  #pragma unroll
  for (int tt = 0; tt < 4; ++tt) {
    int n = n0 + tt * 16 + (lane & 15);
    #pragma unroll
    for (int r = 0; r < 8; ++r) {
      int m = m_blk * 16 + r + ((lane >> 4) << 3);
      out[(size_t)m * DOUT_ + n] = acc[tt][r] + bout[n];
    }
  }
}

// ------ weight conversion: Wcat gate-interleaved (p = 4*j + gate), fp32 -> bf16 ------
__global__ __launch_bounds__(256) void dnc_convert(
    const float* __restrict__ W_ih, const float* __restrict__ W_hh,
    const float* __restrict__ b_ih, const float* __restrict__ b_hh,
    const float* __restrict__ W_int, const float* __restrict__ b_int,
    const float* __restrict__ W_out,
    __bf16* __restrict__ Wcat, __bf16* __restrict__ Wint, __bf16* __restrict__ Wout,
    float* __restrict__ bcomb, float* __restrict__ bint)
{
  const int stride = gridDim.x * blockDim.x;
  const int t0 = blockIdx.x * blockDim.x + threadIdx.x;
  for (int i = t0; i < NGATES * KGATES; i += stride) {
    int p = i >> 10, k = i & 1023;
    int j = p >> 2, gate = p & 3;
    int orig = gate * H_ + j;
    float v = (k < 512) ? W_ih[orig * 512 + k] : W_hh[orig * 512 + (k - 512)];
    Wcat[i] = f2bf(v);
  }
  for (int i = t0; i < NIF * KIF; i += stride) {
    int n = i >> 9;
    Wint[i] = f2bf(n < IFACE_REAL ? W_int[i] : 0.f);
  }
  for (int i = t0; i < NOUT * KOUT; i += stride) Wout[i] = f2bf(W_out[i]);
  for (int p = t0; p < NGATES; p += stride) {
    int j = p >> 2, gate = p & 3;
    int orig = gate * H_ + j;
    bcomb[p] = b_ih[orig] + b_hh[orig];
  }
  for (int i = t0; i < NIF; i += stride) bint[i] = (i < IFACE_REAL) ? b_int[i] : 0.f;
}

__global__ __launch_bounds__(256) void dnc_zero(float* __restrict__ p, int n) {
  for (int i = blockIdx.x * 256 + threadIdx.x; i < n; i += gridDim.x * 256) p[i] = 0.f;
}

extern "C" void kernel_launch(void* const* d_in, const int* in_sizes, int n_in,
                              void* d_out, int out_size, void* d_ws, size_t ws_size,
                              hipStream_t stream) {
  const float* x     = (const float*)d_in[0];
  const float* W_ih  = (const float*)d_in[1];
  const float* W_hh  = (const float*)d_in[2];
  const float* b_ih  = (const float*)d_in[3];
  const float* b_hh  = (const float*)d_in[4];
  const float* W_int = (const float*)d_in[5];
  const float* b_int = (const float*)d_in[6];
  const float* W_out = (const float*)d_in[7];
  const float* b_out = (const float*)d_in[8];
  float* out = (float*)d_out;

  char* ws = (char*)d_ws;
  auto align256 = [](size_t v) { return (v + 255) & ~size_t(255); };
  size_t off = 0;
  __bf16* Wcat = (__bf16*)(ws + off); off = align256(off + (size_t)NGATES * KGATES * 2);
  __bf16* Wint = (__bf16*)(ws + off); off = align256(off + (size_t)NIF * KIF * 2);
  __bf16* Wout = (__bf16*)(ws + off); off = align256(off + (size_t)NOUT * KOUT * 2);
  float* bcomb = (float*)(ws + off);  off = align256(off + (size_t)NGATES * 4);
  float* bint  = (float*)(ws + off);  off = align256(off + (size_t)NIF * 4);
  float* hA    = (float*)(ws + off);  off = align256(off + (size_t)B_ * H_ * 4);
  float* hB    = (float*)(ws + off);  off = align256(off + (size_t)B_ * H_ * 4);
  float* c     = (float*)(ws + off);  off = align256(off + (size_t)B_ * H_ * 4);
  float* mem   = (float*)(ws + off);  off = align256(off + (size_t)B_ * NSLOT_ * WW_ * 4);
  float* rv    = (float*)(ws + off);  off = align256(off + (size_t)B_ * R_ * WW_ * 4);
  float* iface = (float*)(ws + off);  off = align256(off + (size_t)B_ * NIF * 4);

  const size_t a_lds     = (size_t)B_ * AP * sizeof(__bf16);              // 33,792 B
  const size_t gates_lds = a_lds + (size_t)B_ * GP * sizeof(float);       // 67,584 B

  dnc_convert<<<1024, 256, 0, stream>>>(W_ih, W_hh, b_ih, b_hh, W_int, b_int, W_out,
                                        Wcat, Wint, Wout, bcomb, bint);
  dnc_zero<<<512, 256, 0, stream>>>(hA, B_ * H_);
  dnc_zero<<<512, 256, 0, stream>>>(c, B_ * H_);
  dnc_zero<<<512, 256, 0, stream>>>(mem, B_ * NSLOT_ * WW_);
  dnc_zero<<<512, 256, 0, stream>>>(rv, B_ * R_ * WW_);

  for (int t = 0; t < T_; ++t) {
    const float* h_in = (t & 1) ? hB : hA;
    float*       h_out = (t & 1) ? hA : hB;
    dnc_gates_lstm<<<NGATES / 128, 256, gates_lds, stream>>>(x, t, h_in, rv, Wcat, bcomb, c, h_out);
    dnc_iface_gemm<<<NIF / 128, 256, a_lds, stream>>>(h_out, Wint, bint, iface);
    dnc_memory_step<<<B_, 128, 0, stream>>>(iface, mem, rv);
  }
  const float* hfin = ((T_ - 1) & 1) ? hA : hB;
  dnc_out_gemm<<<NOUT / 128, 256, a_lds, stream>>>(hfin, rv, Wout, b_out, out);
}
